// TransformerDecoder_32633161515210
// MI455X (gfx1250) — matrix operand order repaired and checked by reading, not compiled, not run on hardware
//
#include <hip/hip_runtime.h>

// ---------------------------------------------------------------------------
// DETR-style dynamic-conv decoder for MI455X (gfx1250, wave32, WMMA).
// Heavy math per layer:
//   GEMM1: kern = Qs @ fc_w^T : [208x256]x[256x65536] -> v_wmma_f32_16x16x32_bf16
//          (fc_w streamed exactly once: each wave owns an n-strip and holds
//           all 13 M-tile accumulators in VGPRs; A fragments preloaded per
//           k-chunk so loads overlap the WMMA pipe)
//   GEMM2: y = kern @ roi : 200 x ([256x256]x[256x49]) -> WMMA bf16,
//          x^T staged in LDS via global_load_async_to_lds_b128 (ASYNCcnt)
// Small ops (attention E=260, FFN 49->98->98->1, MLP 256->256->4) -> VALU.
// ---------------------------------------------------------------------------

namespace {
constexpr int D    = 256;
constexpr int NQ   = 100;
constexpr int E    = 260;
constexpr int NH   = 4;
constexpr int DH   = 65;            // E / NH
constexpr int P    = 7;
constexpr int NL   = 6;
constexpr int NCLS = 92;
constexpr int BS   = 2;
constexpr int FH   = 32, FW = 32;
constexpr int NROW = BS * NQ;       // 200
constexpr int MPAD = 208;           // NROW padded to 16
constexpr int MT   = MPAD / 16;     // 13 M tiles
constexpr int NK   = D * D;         // 65536 kern columns
constexpr int SP   = 64;            // 49 spatial taps padded to 64
constexpr float LNEPS = 1e-5f;
}

typedef __attribute__((ext_vector_type(16))) __bf16       bf16x16;
typedef __attribute__((ext_vector_type(8)))  float        f32x8;
typedef __attribute__((ext_vector_type(4)))  float        f32x4;
typedef __attribute__((ext_vector_type(4)))  unsigned int u32x4;

union FragB16 {
  __bf16  h[16];
  u32x4   q[2];
  bf16x16 v;
};

__device__ inline f32x8 wmma_bf16(const FragB16& a, const FragB16& b, f32x8 c) {
  return __builtin_amdgcn_wmma_f32_16x16x32_bf16(false, a.v, false, b.v,
                                                 (short)0, c, false, false);
}

// ---------------------------------------------------------------------------
// init: obj <- query_embed, bbox <- bbox_embed, zero A-operand pad rows
// ---------------------------------------------------------------------------
__global__ void k_init(const float* __restrict__ qe, const float* __restrict__ be,
                       float* __restrict__ obj, float* __restrict__ bbox,
                       __bf16* __restrict__ Abf) {
  int t = blockIdx.x * blockDim.x + threadIdx.x;      // 0 .. MPAD*D
  if (t < NROW * D) {
    int row = t / D, d = t % D, q = row % NQ;
    obj[t] = qe[q * D + d];
  }
  if (t < NROW * 4) {
    int row = t / 4, e = t % 4, q = row % NQ;
    bbox[t] = be[q * 4 + e];
  }
  if (t >= NROW * D && t < MPAD * D) Abf[t] = (__bf16)0.f;   // zero padded A rows
}

// ---------------------------------------------------------------------------
// QKV projection: qk = [obj ++ bbox] (E=260), v-input = [obj ++ 0000]
// qkvh[row][0:260]=q*scale, [260:520]=k, [520:780]=v
// ---------------------------------------------------------------------------
__global__ __launch_bounds__(256) void k_attn_qkv(const float* __restrict__ obj,
                                                  const float* __restrict__ bbox,
                                                  const float* __restrict__ w,
                                                  const float* __restrict__ bvec,
                                                  float* __restrict__ qkvh) {
  __shared__ float qk[E];
  const int row = blockIdx.x, t = threadIdx.x;
  if (t < D) qk[t] = obj[row * D + t];
  if (t < 4) qk[D + t] = bbox[row * 4 + t];
  __syncthreads();
  const float scale = 0.12403473458920847f;           // (E/NH)^-0.5 = 65^-0.5
  for (int j = t; j < 3 * E; j += 256) {
    float acc = bvec[j];
    const bool isV = (j >= 2 * E);
    const float* wr = w + (size_t)j * E;
    for (int e = 0; e < E; ++e) {
      float x = qk[e];
      if (isV && e >= D) x = 0.f;                     // F.pad(obj,(0,4))
      acc += wr[e] * x;
    }
    if (j < E) acc *= scale;
    qkvh[(size_t)row * (3 * E) + j] = acc;
  }
}

// ---------------------------------------------------------------------------
// attention core: one block per (batch, head); softmax over 100 keys
// ---------------------------------------------------------------------------
__global__ __launch_bounds__(256) void k_attn_core(const float* __restrict__ qkvh,
                                                   float* __restrict__ atto) {
  const int n = blockIdx.x / NH, h = blockIdx.x % NH;
  __shared__ float kh[NQ * DH];
  __shared__ float vh[NQ * DH];
  __shared__ float qrow[DH];
  __shared__ float sc[NQ];
  __shared__ float red[2];
  const int t = threadIdx.x;
  for (int i = t; i < NQ * DH; i += 256) {
    int q = i / DH, d = i % DH;
    const float* base = qkvh + (size_t)(n * NQ + q) * (3 * E) + h * DH + d;
    kh[i] = base[E];
    vh[i] = base[2 * E];
  }
  __syncthreads();
  for (int q = 0; q < NQ; ++q) {
    if (t < DH) qrow[t] = qkvh[(size_t)(n * NQ + q) * (3 * E) + h * DH + t];
    __syncthreads();
    if (t < NQ) {
      float s = 0.f;
      for (int d = 0; d < DH; ++d) s += qrow[d] * kh[t * DH + d];
      sc[t] = s;
    }
    __syncthreads();
    if (t == 0) {
      float mx = sc[0];
      for (int m = 1; m < NQ; ++m) mx = fmaxf(mx, sc[m]);
      red[0] = mx;
    }
    __syncthreads();
    if (t < NQ) sc[t] = __expf(sc[t] - red[0]);
    __syncthreads();
    if (t == 0) {
      float sm = 0.f;
      for (int m = 0; m < NQ; ++m) sm += sc[m];
      red[1] = 1.f / sm;
    }
    __syncthreads();
    if (t < DH) {
      float o = 0.f;
      for (int m = 0; m < NQ; ++m) o += sc[m] * vh[m * DH + t];
      atto[(size_t)(n * NQ + q) * E + h * DH + t] = o * red[1];
    }
    __syncthreads();
  }
}

// ---------------------------------------------------------------------------
// attention out-proj (keep first 256 dims) + residual + LayerNorm,
// emitting the GEMM1 A-operand directly as bf16.
// ---------------------------------------------------------------------------
__global__ __launch_bounds__(256) void k_attn_out_ln(const float* __restrict__ atto,
                                                     const float* __restrict__ wout,
                                                     const float* __restrict__ bout,
                                                     const float* __restrict__ gam,
                                                     const float* __restrict__ bet,
                                                     const float* __restrict__ obj,
                                                     __bf16* __restrict__ Abf) {
  __shared__ float f[E];
  __shared__ float s1[256];
  const int row = blockIdx.x, t = threadIdx.x;
  for (int i = t; i < E; i += 256) f[i] = atto[(size_t)row * E + i];
  __syncthreads();
  float acc = bout[t];
  const float* wr = wout + (size_t)t * E;
  for (int j = 0; j < E; ++j) acc += wr[j] * f[j];
  const float res = obj[row * D + t] + acc;
  s1[t] = res; __syncthreads();
  for (int o = 128; o; o >>= 1) { if (t < o) s1[t] += s1[t + o]; __syncthreads(); }
  const float mean = s1[0] * (1.f / D); __syncthreads();
  const float dv = res - mean;
  s1[t] = dv * dv; __syncthreads();
  for (int o = 128; o; o >>= 1) { if (t < o) s1[t] += s1[t + o]; __syncthreads(); }
  const float rstd = rsqrtf(s1[0] * (1.f / D) + LNEPS);
  Abf[row * D + t] = (__bf16)(dv * rstd * gam[t] + bet[t]);
}

// ---------------------------------------------------------------------------
// roi_pool: block per (b,q), thread = channel; writes x^T as [s][c] bf16
// (the [N][K] K-contiguous layout the WMMA B fragment wants), zero-padded
// to SP=64 spatial rows.
// ---------------------------------------------------------------------------
__global__ __launch_bounds__(256) void k_roi(const float* __restrict__ feat,
                                             const float* __restrict__ bbox,
                                             __bf16* __restrict__ xT) {
  const int row = blockIdx.x, b = row / NQ, t = threadIdx.x;
  const float bb0 = bbox[row * 4 + 0], bb1 = bbox[row * 4 + 1];
  const float bb2 = bbox[row * 4 + 2], bb3 = bbox[row * 4 + 3];
  const float cx = 1.f / (1.f + __expf(-bb0));
  const float cy = 1.f / (1.f + __expf(-bb1));
  const float ww = 1.f / (1.f + __expf(-bb2));
  const float hh = 1.f / (1.f + __expf(-bb3));
  const float x1 = rintf(cx - 0.5f * ww), y1 = rintf(cy - 0.5f * hh);
  const float x2 = rintf(cx + 0.5f * ww), y2 = rintf(cy + 0.5f * hh);
  const float bw = fmaxf(x2 - x1 + 1.f, 1.f) * (1.f / P);
  const float bh = fmaxf(y2 - y1 + 1.f, 1.f) * (1.f / P);
  const float* fc = feat + ((size_t)b * D + t) * FH * FW;
  for (int ph = 0; ph < P; ++ph) {
    const int hs = (int)fminf(fmaxf(floorf(ph * bh) + y1, 0.f), (float)FH);
    const int he = (int)fminf(fmaxf(ceilf((ph + 1) * bh) + y1, 0.f), (float)FH);
    for (int pw = 0; pw < P; ++pw) {
      const int ws = (int)fminf(fmaxf(floorf(pw * bw) + x1, 0.f), (float)FW);
      const int we = (int)fminf(fmaxf(ceilf((pw + 1) * bw) + x1, 0.f), (float)FW);
      float m = -3.0e38f;
      for (int y = hs; y < he; ++y)
        for (int x = ws; x < we; ++x) m = fmaxf(m, fc[y * FW + x]);
      const float outv = (he > hs && we > ws) ? m : 0.f;
      xT[((size_t)row * SP + ph * P + pw) * D + t] = (__bf16)outv;
    }
  }
  for (int s = P * P; s < SP; ++s) xT[((size_t)row * SP + s) * D + t] = (__bf16)0.f;
}

// ---------------------------------------------------------------------------
// GEMM1: kern[m][n] = sum_d A[m][d] * fc_w[n][d] + fc_b[n]   (bf16 out)
// Each wave owns one 16-wide n-strip and ALL 13 M-tiles (104 acc VGPRs),
// so fc_w (the 67MB stream) is loaded exactly once. Per k-chunk all 13 A
// fragments are preloaded (26-load clause, staggered waits) before the 13
// back-to-back WMMAs, keeping the matrix pipe fed. B converted f32->bf16
// with v_cvt_pk_bf16_f32.
// ---------------------------------------------------------------------------
__global__ __launch_bounds__(256) void k_gemm1(const __bf16* __restrict__ A,
                                               const float* __restrict__ fcw,
                                               const float* __restrict__ fcb,
                                               __bf16* __restrict__ kern) {
  const int wid = threadIdx.x >> 5;
  const int lane = threadIdx.x & 31;
  const int lr = lane & 15, g = lane >> 4;
  const int n0 = (blockIdx.x * 8 + wid) * 16;
  const float* brow = fcw + (size_t)(n0 + lr) * D;
  f32x8 acc[MT];
#pragma unroll
  for (int m = 0; m < MT; ++m) acc[m] = (f32x8){0.f,0.f,0.f,0.f,0.f,0.f,0.f,0.f};
  const char* abase = (const char*)(A + (size_t)lr * D);
#pragma unroll
  for (int k0 = 0; k0 < D; k0 += 32) {
    // preload all 13 A fragments for this k-chunk (independent dests)
    FragB16 afr[MT];
#pragma unroll
    for (int m = 0; m < MT; ++m) {
      const char* ar = abase + (size_t)m * 16 * D * 2;
      afr[m].q[0] = *(const u32x4*)(ar + (k0 * 2 + g * 32));
      afr[m].q[1] = *(const u32x4*)(ar + (k0 * 2 + g * 32 + 16));
    }
    const float* bp = brow + k0 + g * 16;
    __builtin_prefetch(bp + 32, 0, 3);                // next K-chunk prefetch
    f32x4 f0 = *(const f32x4*)(bp + 0);
    f32x4 f1 = *(const f32x4*)(bp + 4);
    f32x4 f2 = *(const f32x4*)(bp + 8);
    f32x4 f3 = *(const f32x4*)(bp + 12);
    FragB16 b;
    b.h[0]  = (__bf16)f0[0]; b.h[1]  = (__bf16)f0[1]; b.h[2]  = (__bf16)f0[2]; b.h[3]  = (__bf16)f0[3];
    b.h[4]  = (__bf16)f1[0]; b.h[5]  = (__bf16)f1[1]; b.h[6]  = (__bf16)f1[2]; b.h[7]  = (__bf16)f1[3];
    b.h[8]  = (__bf16)f2[0]; b.h[9]  = (__bf16)f2[1]; b.h[10] = (__bf16)f2[2]; b.h[11] = (__bf16)f2[3];
    b.h[12] = (__bf16)f3[0]; b.h[13] = (__bf16)f3[1]; b.h[14] = (__bf16)f3[2]; b.h[15] = (__bf16)f3[3];
#pragma unroll
    for (int m = 0; m < MT; ++m) acc[m] = wmma_bf16(afr[m], b, acc[m]);
  }
  const int col = n0 + lr;
  const float bias = fcb[col];
#pragma unroll
  for (int m = 0; m < MT; ++m) {
#pragma unroll
    for (int r = 0; r < 8; ++r) {
      const int mr = m * 16 + r + 8 * g;              // C layout: M=r+8g, N=lane&15
      kern[(size_t)mr * NK + col] = (__bf16)(acc[m][r] + bias);
    }
  }
}

// ---------------------------------------------------------------------------
// GEMM2: per (b,q): y[o][s] = sum_c kern[o][c] * xT[s][c], fused BatchNorm.
// x^T (64x256 bf16, 32KB) staged in LDS with async global->LDS copies
// (ASYNCcnt path); 8 waves x (2 Mtiles x 4 Ntiles) of WMMA. A fragments for
// all 8 k-steps are preloaded from global before the WMMA loop, so only LDS
// ds_read_b128s remain inside.
// ---------------------------------------------------------------------------
__global__ __launch_bounds__(256) void k_gemm2(const __bf16* __restrict__ kern,
                                               const __bf16* __restrict__ xT,
                                               const float* __restrict__ bng,
                                               const float* __restrict__ bnb,
                                               const float* __restrict__ bnm,
                                               const float* __restrict__ bnv,
                                               float* __restrict__ y) {
  __shared__ __align__(16) __bf16 xs[SP * D];                  // 32 KB
  const int row = blockIdx.x, t = threadIdx.x;
  {
    const __bf16* src = xT + (size_t)row * SP * D;
    for (int i = t; i < SP * D / 8; i += 256) {                // 16B chunks
      unsigned lds = (unsigned)(unsigned long long)(const void*)(xs + i * 8);
      const void* gp = (const void*)(src + i * 8);
      asm volatile("global_load_async_to_lds_b128 %0, %1, off"
                   :: "v"(lds), "v"(gp) : "memory");
    }
    asm volatile("s_wait_asynccnt 0" ::: "memory");
  }
  __syncthreads();
  const int wid = t >> 5, lane = t & 31;
  const int lr = lane & 15, g = lane >> 4;
#pragma unroll
  for (int mi = 0; mi < 2; ++mi) {
    const int mt = wid * 2 + mi;
    const char* arow = (const char*)(kern + (size_t)row * NK + (size_t)(mt * 16 + lr) * D);
    // preload all 8 k-step A fragments (global, independent dests)
    FragB16 afr[8];
#pragma unroll
    for (int ks = 0; ks < 8; ++ks) {
      afr[ks].q[0] = *(const u32x4*)(arow + (ks * 64 + g * 16));
      afr[ks].q[1] = *(const u32x4*)(arow + (ks * 64 + 32 + g * 16));
    }
    f32x8 acc[4];
#pragma unroll
    for (int nt = 0; nt < 4; ++nt) acc[nt] = (f32x8){0.f,0.f,0.f,0.f,0.f,0.f,0.f,0.f};
#pragma unroll
    for (int ks = 0; ks < 8; ++ks) {
      const int k0 = ks * 32;
#pragma unroll
      for (int nt = 0; nt < 4; ++nt) {
        const char* bb = (const char*)xs + (nt * 16 + lr) * (D * 2);
        FragB16 b;
        b.q[0] = *(const u32x4*)(bb + (k0 * 2 + g * 16));
        b.q[1] = *(const u32x4*)(bb + (k0 * 2 + 32 + g * 16));
        acc[nt] = wmma_bf16(afr[ks], b, acc[nt]);
      }
    }
#pragma unroll
    for (int r = 0; r < 8; ++r) {
      const int o = mt * 16 + r + 8 * g;
      const float inv = bng[o] * rsqrtf(bnv[o] + LNEPS);
      const float bt2 = bnb[o] - bnm[o] * inv;
      float* yr = y + ((size_t)row * D + o) * SP;
#pragma unroll
      for (int nt = 0; nt < 4; ++nt) yr[nt * 16 + lr] = acc[nt][r] * inv + bt2;
    }
  }
}

// ---------------------------------------------------------------------------
// spatial FFN: per (b,q,o): t=relu(y@w1^T+b1); t=relu(t@w2^T+b2); t@w3^T+b3
// block per (b,q); each wave owns 32 output channels, per-wave LDS scratch.
// ---------------------------------------------------------------------------
__global__ __launch_bounds__(256) void k_ffn(const float* __restrict__ y,
                                             const float* __restrict__ w1,
                                             const float* __restrict__ b1,
                                             const float* __restrict__ w2,
                                             const float* __restrict__ b2,
                                             const float* __restrict__ w3,
                                             const float* __restrict__ b3,
                                             float* __restrict__ objNew) {
  __shared__ float sw2[98 * 98];
  __shared__ float sb1[98], sb2[98], sw3[98];
  __shared__ float ys[8][49], t1s[8][98], t2s[8][98];
  const int row = blockIdx.x, t = threadIdx.x;
  for (int i = t; i < 98 * 98; i += 256) sw2[i] = w2[i];
  if (t < 98) { sb1[t] = b1[t]; sb2[t] = b2[t]; sw3[t] = w3[t]; }
  __syncthreads();
  const int wid = t >> 5, lane = t & 31;
  const float b3v = b3[0];
  for (int r = 0; r < 32; ++r) {
    const int o = wid * 32 + r;
    const float* yrow = y + ((size_t)row * D + o) * SP;
    ys[wid][lane < 49 ? lane : 0] = yrow[lane < 49 ? lane : 0];
    if (lane + 32 < 49) ys[wid][lane + 32] = yrow[lane + 32];
    __syncthreads();
    for (int j = lane; j < 98; j += 32) {
      float a = sb1[j];
      const float* wr = w1 + j * 49;
      for (int i = 0; i < 49; ++i) a += wr[i] * ys[wid][i];
      t1s[wid][j] = fmaxf(a, 0.f);
    }
    __syncthreads();
    for (int j = lane; j < 98; j += 32) {
      float a = sb2[j];
      const float* wr = sw2 + j * 98;
      for (int i = 0; i < 98; ++i) a += wr[i] * t1s[wid][i];
      t2s[wid][j] = fmaxf(a, 0.f);
    }
    __syncthreads();
    float a = 0.f;
    for (int j = lane; j < 98; j += 32) a += sw3[j] * t2s[wid][j];
    for (int off = 16; off; off >>= 1) a += __shfl_down(a, off, 32);
    if (lane == 0) objNew[row * D + o] = a + b3v;
    __syncthreads();
  }
}

// ---------------------------------------------------------------------------
// bbox MLP: new_b = lin3(relu(lin2(relu(lin1(new_q)))))
// ---------------------------------------------------------------------------
__global__ __launch_bounds__(256) void k_lin2(const float* __restrict__ obj,
                                              const float* __restrict__ w1,
                                              const float* __restrict__ b1,
                                              const float* __restrict__ w2,
                                              const float* __restrict__ b2,
                                              const float* __restrict__ w3,
                                              const float* __restrict__ b3,
                                              float* __restrict__ bbox) {
  __shared__ float x[D], u1[D], u2[D];
  const int row = blockIdx.x, t = threadIdx.x;
  x[t] = obj[row * D + t];
  __syncthreads();
  float a = b1[t];
  { const float* wr = w1 + (size_t)t * D; for (int i = 0; i < D; ++i) a += wr[i] * x[i]; }
  u1[t] = fmaxf(a, 0.f);
  __syncthreads();
  a = b2[t];
  { const float* wr = w2 + (size_t)t * D; for (int i = 0; i < D; ++i) a += wr[i] * u1[i]; }
  u2[t] = fmaxf(a, 0.f);
  __syncthreads();
  if (t < 4) {
    a = b3[t];
    const float* wr = w3 + (size_t)t * D;
    for (int i = 0; i < D; ++i) a += wr[i] * u2[i];
    bbox[row * 4 + t] = a;
  }
}

// ---------------------------------------------------------------------------
// head: final LayerNorm + class logits + raw boxes -> d_out
// d_out = logits [bs*NQ*92] ++ boxes [bs*NQ*4]
// ---------------------------------------------------------------------------
__global__ __launch_bounds__(256) void k_head(const float* __restrict__ obj,
                                              const float* __restrict__ bbox,
                                              const float* __restrict__ gam,
                                              const float* __restrict__ bet,
                                              const float* __restrict__ clsw,
                                              const float* __restrict__ clsb,
                                              float* __restrict__ out) {
  __shared__ float s1[256];
  __shared__ float nr[D];
  const int row = blockIdx.x, t = threadIdx.x;
  const float v = obj[row * D + t];
  s1[t] = v; __syncthreads();
  for (int o = 128; o; o >>= 1) { if (t < o) s1[t] += s1[t + o]; __syncthreads(); }
  const float mean = s1[0] * (1.f / D); __syncthreads();
  const float dv = v - mean;
  s1[t] = dv * dv; __syncthreads();
  for (int o = 128; o; o >>= 1) { if (t < o) s1[t] += s1[t + o]; __syncthreads(); }
  const float rstd = rsqrtf(s1[0] * (1.f / D) + LNEPS);
  nr[t] = dv * rstd * gam[t] + bet[t];
  __syncthreads();
  if (t < NCLS) {
    float a = clsb[t];
    const float* wr = clsw + (size_t)t * D;
    for (int i = 0; i < D; ++i) a += wr[i] * nr[i];
    out[(size_t)row * NCLS + t] = a;
  }
  if (t < 4) out[(size_t)BS * NQ * NCLS + row * 4 + t] = bbox[row * 4 + t];
}

// ---------------------------------------------------------------------------
// host
// ---------------------------------------------------------------------------
extern "C" void kernel_launch(void* const* d_in, const int* in_sizes, int n_in,
                              void* d_out, int out_size, void* d_ws, size_t ws_size,
                              hipStream_t stream) {
  (void)in_sizes; (void)n_in; (void)out_size; (void)ws_size;
  const float* feat = (const float*)d_in[0];
  auto LP = [&](int l, int k) -> const float* { return (const float*)d_in[1 + l * 24 + k]; };
  const int pb = 1 + NL * 24;
  const float* query_embed = (const float*)d_in[pb + 0];
  const float* bbox_embed  = (const float*)d_in[pb + 1];
  const float* norm_g      = (const float*)d_in[pb + 2];
  const float* norm_b      = (const float*)d_in[pb + 3];
  const float* cls_w       = (const float*)d_in[pb + 4];
  const float* cls_b       = (const float*)d_in[pb + 5];

  char* wsp = (char*)d_ws;
  size_t off = 0;
  auto alloc = [&](size_t bytes) -> void* {
    void* p = wsp + off;
    off = (off + bytes + 255) & ~(size_t)255;
    return p;
  };
  float*  obj  = (float*)alloc((size_t)NROW * D * 4);
  float*  bbox = (float*)alloc((size_t)NROW * 4 * 4);
  float*  qkvh = (float*)alloc((size_t)NROW * 3 * E * 4);
  float*  atto = (float*)alloc((size_t)NROW * E * 4);
  __bf16* Abf  = (__bf16*)alloc((size_t)MPAD * D * 2);
  __bf16* xT   = (__bf16*)alloc((size_t)NROW * SP * D * 2);
  __bf16* kern = (__bf16*)alloc((size_t)MPAD * NK * 2);
  float*  yb   = (float*)alloc((size_t)NROW * D * SP * 4);
  // total workspace ~48.1 MB

  k_init<<<(MPAD * D + 255) / 256, 256, 0, stream>>>(query_embed, bbox_embed, obj, bbox, Abf);

  for (int l = 0; l < NL; ++l) {
    k_attn_qkv<<<NROW, 256, 0, stream>>>(obj, bbox, LP(l, 0), LP(l, 1), qkvh);
    k_attn_core<<<BS * NH, 256, 0, stream>>>(qkvh, atto);
    k_attn_out_ln<<<NROW, 256, 0, stream>>>(atto, LP(l, 2), LP(l, 3),
                                            LP(l, 12), LP(l, 13), obj, Abf);
    k_roi<<<NROW, 256, 0, stream>>>(feat, bbox, xT);
    k_gemm1<<<NK / 128, 256, 0, stream>>>(Abf, LP(l, 4), LP(l, 5), kern);
    k_gemm2<<<NROW, 256, 0, stream>>>(kern, xT, LP(l, 20), LP(l, 21),
                                      LP(l, 22), LP(l, 23), yb);
    k_ffn<<<NROW, 256, 0, stream>>>(yb, LP(l, 6), LP(l, 7), LP(l, 8), LP(l, 9),
                                    LP(l, 10), LP(l, 11), obj);
    k_lin2<<<NROW, 256, 0, stream>>>(obj, LP(l, 14), LP(l, 15), LP(l, 16), LP(l, 17),
                                     LP(l, 18), LP(l, 19), bbox);
  }

  k_head<<<NROW, 256, 0, stream>>>(obj, bbox, norm_g, norm_b, cls_w, cls_b, (float*)d_out);
}
